// DeepSeekMoELayer_13683765805415
// MI455X (gfx1250) — compile-verified, matrix operand order checked by reading, not run on hardware
//
#include <hip/hip_runtime.h>
#include <hip/hip_bf16.h>
#include <math.h>

// ---------------------------------------------------------------------------
// DeepSeek-MoE layer for MI455X (gfx1250), wave32.
// GEMMs: v_wmma_f32_16x16x32_bf16, LDS-staged via GLOBAL_LOAD_ASYNC_TO_LDS_B128
// (ASYNCcnt) with double buffering; 128-wide macro tiles shared by 8 waves.
// ---------------------------------------------------------------------------

typedef __attribute__((ext_vector_type(16))) __bf16 v16bf;
typedef __attribute__((ext_vector_type(8)))  float  v8f;

#define D_MODEL   1024
#define N_EXPERTS 8
#define TOP_K_    2
#define N_SHARED_ 2
#define HS        2048   // shared expert hidden
#define HR        1024   // routed expert hidden
#define BB        4
#define TT        2048
#define NTOK      (BB * TT)   // 8192 tokens

// padded LDS row: 32 K-elements + 8 pad = 40 bf16 (80B) -> conflict-free b128 reads
#define ROWEL 40

// ---- workspace layout (bytes) ----
#define OFF_XB    ((size_t)0)                       // x bf16      [NTOK, D_MODEL]
#define OFF_WS1   ((size_t)16777216)                // s_w1 bf16   [2, HS, D_MODEL]
#define OFF_WS3   ((size_t)25165824)                // s_w3 bf16
#define OFF_WS2   ((size_t)33554432)                // s_w2 bf16   [2, D_MODEL, HS]
#define OFF_WR1   ((size_t)41943040)                // r_w1 bf16   [8, HR, D_MODEL]
#define OFF_WR2   ((size_t)58720256)                // r_w2 bf16   [8, D_MODEL, HR]
#define OFF_GATES ((size_t)75497472)                // gates f32   [NTOK, 8]
#define OFF_HBUF  ((size_t)75759616)                // h bf16      [NTOK, HS] (reused)

// ---------------------------------------------------------------------------
// Async copy of one KxROWS tile (rows x 32 K bf16) into padded LDS tile.
// Each chunk = 16B (8 bf16); ROWS*4 chunks spread over 256 threads.
// ---------------------------------------------------------------------------
template <int NROWS>
__device__ __forceinline__ void issue_tile_async(const __bf16* __restrict__ g,
                                                 int gstride, int row0, int kb,
                                                 __bf16* lds, int tid) {
#pragma unroll
  for (int q = 0; q < (NROWS * 4) / 256; ++q) {
    int c = tid + q * 256;
    int r = c >> 2, seg = c & 3;
    const __bf16* gp = g + (size_t)(row0 + r) * (size_t)gstride + kb + seg * 8;
    unsigned long long ga = (unsigned long long)(uintptr_t)gp;
    unsigned int la = (unsigned int)(uintptr_t)(lds + r * ROWEL + seg * 8);
    asm volatile("global_load_async_to_lds_b128 %0, %1, off"
                 :: "v"(la), "v"(ga) : "memory");
  }
#pragma unroll
  for (int q = 0; q < ((NROWS * 4) % 256 ? 1 : 0); ++q) {
    // NROWS*4 < 256 case handled by the generic path below (never both).
  }
  if constexpr ((NROWS * 4) < 256) {
    int c = tid;
    if (c < NROWS * 4) {
      int r = c >> 2, seg = c & 3;
      const __bf16* gp = g + (size_t)(row0 + r) * (size_t)gstride + kb + seg * 8;
      unsigned long long ga = (unsigned long long)(uintptr_t)gp;
      unsigned int la = (unsigned int)(uintptr_t)(lds + r * ROWEL + seg * 8);
      asm volatile("global_load_async_to_lds_b128 %0, %1, off"
                   :: "v"(la), "v"(ga) : "memory");
    }
  }
}

__device__ __forceinline__ void wait_async_all() {
  asm volatile("s_wait_asynccnt 0x0" ::: "memory");
}

// Fragment read from padded LDS tile: two 16B ds loads per lane.
__device__ __forceinline__ v16bf load_frag_lds(const __bf16* p) {
  union { v16bf v; uint4 q[2]; } u;
  u.q[0] = *reinterpret_cast<const uint4*>(p);
  u.q[1] = *reinterpret_cast<const uint4*>(p + 16);
  return u.v;
}

__device__ __forceinline__ v8f wmma_bf16(v16bf a, v16bf b, v8f c) {
  return __builtin_amdgcn_wmma_f32_16x16x32_bf16(false, a, false, b, (short)0, c,
                                                 false, false);
}

// ---------------------------------------------------------------------------
// fp32 -> bf16 conversion / zero (grid-stride)
// ---------------------------------------------------------------------------
__global__ void cvt_bf16_kernel(const float* __restrict__ in,
                                __bf16* __restrict__ out, size_t n) {
  size_t i = (size_t)blockIdx.x * blockDim.x + threadIdx.x;
  size_t stride = (size_t)gridDim.x * blockDim.x;
  for (; i < n; i += stride) out[i] = (__bf16)in[i];
}

__global__ void zero_f32_kernel(float* __restrict__ p, size_t n) {
  size_t i = (size_t)blockIdx.x * blockDim.x + threadIdx.x;
  size_t stride = (size_t)gridDim.x * blockDim.x;
  for (; i < n; i += stride) p[i] = 0.0f;
}

// ---------------------------------------------------------------------------
// Router: one wave32 per token -> dense gate table gates[NTOK][8].
// ---------------------------------------------------------------------------
__global__ __launch_bounds__(256) void router_kernel(
    const float* __restrict__ x, const float* __restrict__ t_emb,
    const float* __restrict__ Wr, const float* __restrict__ bias,
    float* __restrict__ gates) {
  const int lane = threadIdx.x & 31;
  const int m = (int)((blockIdx.x * blockDim.x + threadIdx.x) >> 5);
  if (m >= NTOK) return;
  const int b = m / TT;

  float acc[N_EXPERTS];
#pragma unroll
  for (int e = 0; e < N_EXPERTS; ++e) acc[e] = 0.0f;

  for (int k = lane; k < 2 * D_MODEL; k += 32) {
    float xv = (k < D_MODEL) ? x[(size_t)m * D_MODEL + k]
                             : t_emb[(size_t)b * D_MODEL + (k - D_MODEL)];
#pragma unroll
    for (int e = 0; e < N_EXPERTS; ++e) acc[e] += xv * Wr[e * 2 * D_MODEL + k];
  }
#pragma unroll
  for (int e = 0; e < N_EXPERTS; ++e) {
#pragma unroll
    for (int off = 16; off > 0; off >>= 1)
      acc[e] += __shfl_xor(acc[e], off, 32);
  }
  if (lane == 0) {
    float s[N_EXPERTS], sel[N_EXPERTS];
#pragma unroll
    for (int e = 0; e < N_EXPERTS; ++e) {
      s[e] = 1.0f / (1.0f + __expf(-acc[e]));
      sel[e] = s[e] + bias[e];
    }
    int i0 = 0;
#pragma unroll
    for (int e = 1; e < N_EXPERTS; ++e) if (sel[e] > sel[i0]) i0 = e;
    int i1 = (i0 == 0) ? 1 : 0;
#pragma unroll
    for (int e = 0; e < N_EXPERTS; ++e)
      if (e != i0 && sel[e] > sel[i1]) i1 = e;
    float denom = s[i0] + s[i1];
    float g0, g1;
    if (denom > 1e-9f) {
      g0 = s[i0] / (denom + 1e-9f);
      g1 = s[i1] / (denom + 1e-9f);
    } else {
      g0 = g1 = 1.0f / (float)TOP_K_;
    }
#pragma unroll
    for (int e = 0; e < N_EXPERTS; ++e) gates[(size_t)m * N_EXPERTS + e] = 0.0f;
    gates[(size_t)m * N_EXPERTS + i0] = g0;
    gates[(size_t)m * N_EXPERTS + i1] = g1;
  }
}

// ---------------------------------------------------------------------------
// Cooperative WMMA GEMM with async-LDS double buffering.
//   MODE 0: C = A@W1^T, exact-erf GELU -> bf16 Hout           [routed up]
//   MODE 1: Out += scale(m) * (A@W1^T)                        [down accum]
//   MODE 2: Hout = silu(A@W1^T) * (A@W3^T) -> bf16            [shared SwiGLU up]
// Macro tile: BM=128 rows of A; BN=128 (MODE 0/1) or 64 (MODE 2) weight rows.
// 8 waves as 4(M) x 2(N); wave tile 32x64 (MODE 0/1) or 32x32 dual (MODE 2).
// ---------------------------------------------------------------------------
template <int MODE>
__global__ __launch_bounds__(256) void coop_gemm_kernel(
    const __bf16* __restrict__ A, int K, const __bf16* __restrict__ W1,
    const __bf16* __restrict__ W3, int Nn, __bf16* __restrict__ Hout,
    float* __restrict__ Out, const float* __restrict__ rowscale, int rs_stride,
    float const_scale) {
  constexpr int BM = 128;
  constexpr int BN = (MODE == 2) ? 64 : 128;
  constexpr int NFRAG = (MODE == 2) ? 2 : 4;

  __shared__ __bf16 shA[2][BM * ROWEL];
  __shared__ __bf16 shB[2][BN * ROWEL];
  __shared__ __bf16 shB3[2][(MODE == 2) ? (BN * ROWEL) : 8];

  const int tid = threadIdx.x;
  const int lane = tid & 31;
  const int half = lane >> 4;
  const int lr = lane & 15;
  const int w = tid >> 5;
  const int mw = w >> 1;           // 0..3
  const int nw = w & 1;            // 0..1
  const int m0 = mw * 32;          // wave M offset within macro tile
  const int n0 = nw * (BN / 2);    // wave N offset within macro tile

  const int nblk = Nn / BN;
  const int bm = (int)blockIdx.x / nblk;
  const int bn = (int)blockIdx.x % nblk;
  const int mBase = bm * BM;
  const int nBase = bn * BN;

  v8f c1[2][NFRAG];
  v8f c3[2][NFRAG];
#pragma unroll
  for (int i = 0; i < 2; ++i)
#pragma unroll
    for (int j = 0; j < NFRAG; ++j) { c1[i][j] = (v8f){}; c3[i][j] = (v8f){}; }

  // prologue: stage K-step 0 into buffer 0
  issue_tile_async<BM>(A, K, mBase, 0, &shA[0][0], tid);
  issue_tile_async<BN>(W1, K, nBase, 0, &shB[0][0], tid);
  if constexpr (MODE == 2) issue_tile_async<BN>(W3, K, nBase, 0, &shB3[0][0], tid);
  wait_async_all();
  __syncthreads();

  const int nk = K / 32;
  for (int it = 0; it < nk; ++it) {
    const int cur = it & 1;
    const int nxt = cur ^ 1;
    if (it + 1 < nk) {
      const int kb = (it + 1) * 32;
      issue_tile_async<BM>(A, K, mBase, kb, &shA[nxt][0], tid);
      issue_tile_async<BN>(W1, K, nBase, kb, &shB[nxt][0], tid);
      if constexpr (MODE == 2)
        issue_tile_async<BN>(W3, K, nBase, kb, &shB3[nxt][0], tid);
    }

    // consume current buffer
    v16bf a[2];
    a[0] = load_frag_lds(&shA[cur][(m0 + lr) * ROWEL + half * 8]);
    a[1] = load_frag_lds(&shA[cur][(m0 + 16 + lr) * ROWEL + half * 8]);
    if constexpr (MODE == 2) {
      v16bf b1[2], b3[2];
#pragma unroll
      for (int j = 0; j < 2; ++j) {
        b1[j] = load_frag_lds(&shB[cur][(n0 + j * 16 + lr) * ROWEL + half * 8]);
        b3[j] = load_frag_lds(&shB3[cur][(n0 + j * 16 + lr) * ROWEL + half * 8]);
      }
#pragma unroll
      for (int i = 0; i < 2; ++i)
#pragma unroll
        for (int j = 0; j < 2; ++j) {
          c1[i][j] = wmma_bf16(a[i], b1[j], c1[i][j]);
          c3[i][j] = wmma_bf16(a[i], b3[j], c3[i][j]);
        }
    } else {
      v16bf b[4];
#pragma unroll
      for (int j = 0; j < 4; ++j)
        b[j] = load_frag_lds(&shB[cur][(n0 + j * 16 + lr) * ROWEL + half * 8]);
#pragma unroll
      for (int i = 0; i < 2; ++i)
#pragma unroll
        for (int j = 0; j < 4; ++j) c1[i][j] = wmma_bf16(a[i], b[j], c1[i][j]);
    }

    if (it + 1 < nk) {
      wait_async_all();   // my async writes into buf nxt have landed
      __syncthreads();    // everyone's landed; everyone done reading buf cur
    }
  }

  // ---- epilogue ----
  // C layout: VGPR v, lanes 0-15 -> (M=v, N=lane); lanes 16-31 -> (M=v+8, N=lane-16)
  if constexpr (MODE == 2) {
#pragma unroll
    for (int i = 0; i < 2; ++i)
#pragma unroll
      for (int j = 0; j < 2; ++j)
#pragma unroll
        for (int v = 0; v < 8; ++v) {
          int row = mBase + m0 + i * 16 + half * 8 + v;
          int col = nBase + n0 + j * 16 + lr;
          float f1 = c1[i][j][v];
          float f3 = c3[i][j][v];
          float h = (f1 / (1.0f + __expf(-f1))) * f3;  // silu(f1) * f3
          Hout[(size_t)row * Nn + col] = (__bf16)h;
        }
  } else if constexpr (MODE == 0) {
#pragma unroll
    for (int i = 0; i < 2; ++i)
#pragma unroll
      for (int j = 0; j < 4; ++j)
#pragma unroll
        for (int v = 0; v < 8; ++v) {
          int row = mBase + m0 + i * 16 + half * 8 + v;
          int col = nBase + n0 + j * 16 + lr;
          float xx = c1[i][j][v];
          float g = 0.5f * xx * (1.0f + erff(xx * 0.70710678118654752f));
          Hout[(size_t)row * Nn + col] = (__bf16)g;
        }
  } else {
    float rs[2][8];
#pragma unroll
    for (int i = 0; i < 2; ++i)
#pragma unroll
      for (int v = 0; v < 8; ++v) {
        int row = mBase + m0 + i * 16 + half * 8 + v;
        float g = rowscale ? rowscale[(size_t)row * rs_stride] : 1.0f;
        rs[i][v] = const_scale * g;
      }
#pragma unroll
    for (int i = 0; i < 2; ++i)
#pragma unroll
      for (int j = 0; j < 4; ++j)
#pragma unroll
        for (int v = 0; v < 8; ++v) {
          int row = mBase + m0 + i * 16 + half * 8 + v;
          int col = nBase + n0 + j * 16 + lr;
          Out[(size_t)row * D_MODEL + col] += c1[i][j][v] * rs[i][v];
        }
  }
}

// ---------------------------------------------------------------------------
// Host-side launch
// ---------------------------------------------------------------------------
extern "C" void kernel_launch(void* const* d_in, const int* in_sizes, int n_in,
                              void* d_out, int out_size, void* d_ws,
                              size_t ws_size, hipStream_t stream) {
  const float* x        = (const float*)d_in[0];  // [4,2048,1024]
  const float* t_emb    = (const float*)d_in[1];  // [4,1024]
  const float* W_router = (const float*)d_in[2];  // [8,2048]
  const float* r_bias   = (const float*)d_in[3];  // [8]
  const float* s_w1     = (const float*)d_in[4];  // [2,2048,1024]
  const float* s_w3     = (const float*)d_in[5];  // [2,2048,1024]
  const float* s_w2     = (const float*)d_in[6];  // [2,1024,2048]
  const float* r_w1     = (const float*)d_in[7];  // [8,1024,1024]
  const float* r_w2     = (const float*)d_in[8];  // [8,1024,1024]
  float* out = (float*)d_out;                     // [4,2048,1024]

  char* ws = (char*)d_ws;
  __bf16* xb    = (__bf16*)(ws + OFF_XB);
  __bf16* ws1   = (__bf16*)(ws + OFF_WS1);
  __bf16* ws3   = (__bf16*)(ws + OFF_WS3);
  __bf16* ws2   = (__bf16*)(ws + OFF_WS2);
  __bf16* wr1   = (__bf16*)(ws + OFF_WR1);
  __bf16* wr2   = (__bf16*)(ws + OFF_WR2);
  float*  gates = (float*)(ws + OFF_GATES);
  __bf16* hbuf  = (__bf16*)(ws + OFF_HBUF);

  // 1) fp32 -> bf16 conversions
  cvt_bf16_kernel<<<2048, 256, 0, stream>>>(x, xb, (size_t)NTOK * D_MODEL);
  cvt_bf16_kernel<<<2048, 256, 0, stream>>>(s_w1, ws1, (size_t)N_SHARED_ * HS * D_MODEL);
  cvt_bf16_kernel<<<2048, 256, 0, stream>>>(s_w3, ws3, (size_t)N_SHARED_ * HS * D_MODEL);
  cvt_bf16_kernel<<<2048, 256, 0, stream>>>(s_w2, ws2, (size_t)N_SHARED_ * D_MODEL * HS);
  cvt_bf16_kernel<<<2048, 256, 0, stream>>>(r_w1, wr1, (size_t)N_EXPERTS * HR * D_MODEL);
  cvt_bf16_kernel<<<2048, 256, 0, stream>>>(r_w2, wr2, (size_t)N_EXPERTS * D_MODEL * HR);

  // 2) zero output accumulator
  zero_f32_kernel<<<1024, 256, 0, stream>>>(out, (size_t)NTOK * D_MODEL);

  // 3) router gates
  router_kernel<<<NTOK / 8, 256, 0, stream>>>(x, t_emb, W_router, r_bias, gates);

  // 4) shared experts: SwiGLU up (MODE 2) + down accumulate (MODE 1, scale 1/8)
  for (int e = 0; e < N_SHARED_; ++e) {
    coop_gemm_kernel<2><<<(NTOK / 128) * (HS / 64), 256, 0, stream>>>(
        xb, D_MODEL, ws1 + (size_t)e * HS * D_MODEL,
        ws3 + (size_t)e * HS * D_MODEL, HS, hbuf, nullptr, nullptr, 0, 0.0f);
    coop_gemm_kernel<1><<<(NTOK / 128) * (D_MODEL / 128), 256, 0, stream>>>(
        hbuf, HS, ws2 + (size_t)e * D_MODEL * HS, nullptr, D_MODEL, nullptr,
        out, nullptr, 0, 0.125f);
  }

  // 5) routed experts: GELU up (MODE 0) + gated down accumulate (MODE 1)
  for (int e = 0; e < N_EXPERTS; ++e) {
    coop_gemm_kernel<0><<<(NTOK / 128) * (HR / 128), 256, 0, stream>>>(
        xb, D_MODEL, wr1 + (size_t)e * HR * D_MODEL, nullptr, HR, hbuf,
        nullptr, nullptr, 0, 0.0f);
    coop_gemm_kernel<1><<<(NTOK / 128) * (D_MODEL / 128), 256, 0, stream>>>(
        hbuf, HR, wr2 + (size_t)e * D_MODEL * HR, nullptr, D_MODEL, nullptr,
        out, gates + e, N_EXPERTS, 0.25f);
  }
}
